// Graph2Seq_72619307041264
// MI455X (gfx1250) — compile-verified, matrix operand order checked by reading, not run on hardware
//
#include <hip/hip_runtime.h>
#include <hip/hip_bf16.h>

typedef __bf16 bf16;
typedef bf16  v16bf __attribute__((ext_vector_type(16)));
typedef float v8f   __attribute__((ext_vector_type(8)));

__device__ __forceinline__ bf16 f2bf(float f) {
    unsigned u = __builtin_bit_cast(unsigned, f);
    unsigned r = u + 0x7FFFu + ((u >> 16) & 1u);       // round-to-nearest-even
    unsigned short h = (unsigned short)(r >> 16);
    return __builtin_bit_cast(bf16, h);
}

__device__ __forceinline__ unsigned pack2bf(float a, float b) {
    unsigned lo = (unsigned)__builtin_bit_cast(unsigned short, f2bf(a));
    unsigned hi = (unsigned)__builtin_bit_cast(unsigned short, f2bf(b));
    return lo | (hi << 16);
}

__device__ __forceinline__ float sigf(float x) { return 1.0f / (1.0f + __expf(-x)); }

// ---------------------------------------------------------------------------
// Generic bf16-WMMA GEMM:  C = [relu]( A@B  (+ A2@B2)  + bias )
// A: M x K (f32, lda) ; B: K x N (f32, ldb) or N x K if transB ; C: M x N (ldc)
// Tile: 32 (M) x 128 (N) x 32 (K); 256 threads = 8 waves, each wave does a
// 16x16 D-tile pair (two v_wmma_f32_16x16x32_bf16 per k-step, x2 if dual).
// LDS holds A row-major and B TRANSPOSED (k contiguous per column) so every
// lane's fragment is two 16B-aligned contiguous runs -> ds_load_b128.
// ---------------------------------------------------------------------------
#define MT 32
#define NT 128
#define KT 32
#define LDS_K 40          // 80 bytes row stride (multiple of 16B)

__global__ __launch_bounds__(256)
void wmma_gemm(const float* __restrict__ A, int lda,
               const float* __restrict__ B, int ldb, int transB,
               const float* __restrict__ A2, const float* __restrict__ B2,
               const float* __restrict__ bias,
               float* __restrict__ C, int ldc,
               int M, int N, int K, int relu)
{
    __shared__ bf16 As  [MT * LDS_K];     // As[r*LDS_K + k]
    __shared__ bf16 BsT [NT * LDS_K];     // BsT[n*LDS_K + k]
    __shared__ bf16 As2 [MT * LDS_K];
    __shared__ bf16 BsT2[NT * LDS_K];

    const int tid  = threadIdx.x;
    const int wave = tid >> 5;
    const int lane = tid & 31;
    const int wr   = wave >> 2;          // 0..1
    const int wc   = wave & 3;           // 0..3
    const int mb   = blockIdx.y * MT;
    const int nb   = blockIdx.x * NT;
    const int half = lane >> 4;
    const int l16  = lane & 15;
    const bool dual = (A2 != nullptr);

    v8f acc0 = {}; v8f acc1 = {};
    v8f acc0b = {}; v8f acc1b = {};

    for (int kb = 0; kb < K; kb += KT) {
        // ---- stage A tile: MT*KT/2 = 512 bf16-pairs, 2 per thread ----
        #pragma unroll
        for (int i = 0; i < 2; ++i) {
            int idx = tid + i * 256;           // 0..511
            int r   = idx >> 4;                // 0..31
            int kp  = (idx & 15) << 1;         // 0,2,..,30
            int gm = mb + r, gk = kb + kp;
            float v0 = 0.f, v1 = 0.f, w0 = 0.f, w1 = 0.f;
            if (gm < M) {
                if (gk     < K) v0 = A[(size_t)gm * lda + gk];
                if (gk + 1 < K) v1 = A[(size_t)gm * lda + gk + 1];
                if (dual) {
                    if (gk     < K) w0 = A2[(size_t)gm * lda + gk];
                    if (gk + 1 < K) w1 = A2[(size_t)gm * lda + gk + 1];
                }
                if (kb + KT < K)
                    __builtin_prefetch(&A[(size_t)gm * lda + gk + KT], 0, 1);
            }
            *(unsigned*)&As[r * LDS_K + kp] = pack2bf(v0, v1);
            if (dual) *(unsigned*)&As2[r * LDS_K + kp] = pack2bf(w0, w1);
        }
        // ---- stage B tile transposed: NT*KT/2 = 2048 pairs, 8 per thread ----
        #pragma unroll
        for (int i = 0; i < 8; ++i) {
            int idx = tid + i * 256;           // 0..2047
            int n   = idx & 127;
            int kp  = (idx >> 7) << 1;         // 0,2,..,30
            int gn = nb + n, gk = kb + kp;
            float v0 = 0.f, v1 = 0.f, w0 = 0.f, w1 = 0.f;
            if (gn < N) {
                if (transB) {
                    if (gk     < K) v0 = B[(size_t)gn * ldb + gk];
                    if (gk + 1 < K) v1 = B[(size_t)gn * ldb + gk + 1];
                    if (dual) {
                        if (gk     < K) w0 = B2[(size_t)gn * ldb + gk];
                        if (gk + 1 < K) w1 = B2[(size_t)gn * ldb + gk + 1];
                    }
                } else {
                    if (gk     < K) v0 = B[(size_t)gk * ldb + gn];
                    if (gk + 1 < K) v1 = B[(size_t)(gk + 1) * ldb + gn];
                    if (dual) {
                        if (gk     < K) w0 = B2[(size_t)gk * ldb + gn];
                        if (gk + 1 < K) w1 = B2[(size_t)(gk + 1) * ldb + gn];
                    }
                }
            }
            *(unsigned*)&BsT[n * LDS_K + kp] = pack2bf(v0, v1);
            if (dual) *(unsigned*)&BsT2[n * LDS_K + kp] = pack2bf(w0, w1);
        }
        __syncthreads();

        // ---- fragments (ISA 7.12.2): per-lane data = two 16B runs ----
        const int am = wr * 16 + l16;
        const int n0 = wc * 32 + l16;
        const bf16* ap  = &As [am * LDS_K + half * 8];
        const bf16* bp0 = &BsT[n0 * LDS_K + half * 8];
        const bf16* bp1 = &BsT[(n0 + 16) * LDS_K + half * 8];
        v16bf afrag, bfrag0, bfrag1;
        #pragma unroll
        for (int v = 0; v < 8; ++v) {
            int o = (v >> 2) * 16 + (v & 3) * 2;   // grp*16 + win*2
            afrag [2 * v] = ap [o]; afrag [2 * v + 1] = ap [o + 1];
            bfrag0[2 * v] = bp0[o]; bfrag0[2 * v + 1] = bp0[o + 1];
            bfrag1[2 * v] = bp1[o]; bfrag1[2 * v + 1] = bp1[o + 1];
        }
        acc0 = __builtin_amdgcn_wmma_f32_16x16x32_bf16(false, afrag, false, bfrag0,
                                                       (short)0, acc0, false, false);
        acc1 = __builtin_amdgcn_wmma_f32_16x16x32_bf16(false, afrag, false, bfrag1,
                                                       (short)0, acc1, false, false);
        if (dual) {
            const bf16* ap2  = &As2 [am * LDS_K + half * 8];
            const bf16* bq0  = &BsT2[n0 * LDS_K + half * 8];
            const bf16* bq1  = &BsT2[(n0 + 16) * LDS_K + half * 8];
            v16bf afrag2, c0f, c1f;
            #pragma unroll
            for (int v = 0; v < 8; ++v) {
                int o = (v >> 2) * 16 + (v & 3) * 2;
                afrag2[2 * v] = ap2[o]; afrag2[2 * v + 1] = ap2[o + 1];
                c0f   [2 * v] = bq0[o]; c0f   [2 * v + 1] = bq0[o + 1];
                c1f   [2 * v] = bq1[o]; c1f   [2 * v + 1] = bq1[o + 1];
            }
            acc0b = __builtin_amdgcn_wmma_f32_16x16x32_bf16(false, afrag2, false, c0f,
                                                            (short)0, acc0b, false, false);
            acc1b = __builtin_amdgcn_wmma_f32_16x16x32_bf16(false, afrag2, false, c1f,
                                                            (short)0, acc1b, false, false);
        }
        __syncthreads();
    }

    // ---- store (C/D layout: VGPR r -> row r + half*8) ----
    const int row0 = mb + wr * 16 + half * 8;
    const int col0 = nb + wc * 32 + l16;
    #pragma unroll
    for (int r = 0; r < 8; ++r) {
        int row = row0 + r;
        if (row >= M) continue;
        if (col0 < N) {
            float v = acc0[r] + acc0b[r] + (bias ? bias[col0] : 0.f);
            if (relu) v = fmaxf(v, 0.f);
            C[(size_t)row * ldc + col0] = v;
        }
        int col1 = col0 + 16;
        if (col1 < N) {
            float v = acc1[r] + acc1b[r] + (bias ? bias[col1] : 0.f);
            if (relu) v = fmaxf(v, 0.f);
            C[(size_t)row * ldc + col1] = v;
        }
    }
}

// ---------------------------------------------------------------------------
// Edge scatter: aggf[dst] += xg[src];  aggb[src] += xg[dst]   (H = 256)
// ---------------------------------------------------------------------------
__global__ __launch_bounds__(256)
void scatter_edges(const float* __restrict__ xg, const int* __restrict__ ei,
                   float* __restrict__ aggf, float* __restrict__ aggb, int E)
{
    int t = blockIdx.x * 256 + threadIdx.x;      // over E * 64 float4 chunks
    if (t >= E * 64) return;
    int e = t >> 6;
    int c = (t & 63) << 2;
    int s = ei[e];
    int d = ei[E + e];
    const float4 vs = *(const float4*)(xg + (size_t)s * 256 + c);
    const float4 vd = *(const float4*)(xg + (size_t)d * 256 + c);
    float* pf = aggf + (size_t)d * 256 + c;
    float* pb = aggb + (size_t)s * 256 + c;
    __hip_atomic_fetch_add(pf + 0, vs.x, __ATOMIC_RELAXED, __HIP_MEMORY_SCOPE_AGENT);
    __hip_atomic_fetch_add(pf + 1, vs.y, __ATOMIC_RELAXED, __HIP_MEMORY_SCOPE_AGENT);
    __hip_atomic_fetch_add(pf + 2, vs.z, __ATOMIC_RELAXED, __HIP_MEMORY_SCOPE_AGENT);
    __hip_atomic_fetch_add(pf + 3, vs.w, __ATOMIC_RELAXED, __HIP_MEMORY_SCOPE_AGENT);
    __hip_atomic_fetch_add(pb + 0, vd.x, __ATOMIC_RELAXED, __HIP_MEMORY_SCOPE_AGENT);
    __hip_atomic_fetch_add(pb + 1, vd.y, __ATOMIC_RELAXED, __HIP_MEMORY_SCOPE_AGENT);
    __hip_atomic_fetch_add(pb + 2, vd.z, __ATOMIC_RELAXED, __HIP_MEMORY_SCOPE_AGENT);
    __hip_atomic_fetch_add(pb + 3, vd.w, __ATOMIC_RELAXED, __HIP_MEMORY_SCOPE_AGENT);
}

// ---------------------------------------------------------------------------
// Per-column mean / rstd over 25600 rows (512 columns, one block each)
// ---------------------------------------------------------------------------
__global__ __launch_bounds__(256)
void col_stats(const float* __restrict__ h1, float* __restrict__ mean,
               float* __restrict__ rstd, int Mrows)
{
    int col = blockIdx.x;
    float s = 0.f, ss = 0.f;
    for (int r = threadIdx.x; r < Mrows; r += 256) {
        float v = h1[(size_t)r * 512 + col];
        s += v; ss += v * v;
    }
    __shared__ float sh[256], sh2[256];
    sh[threadIdx.x] = s; sh2[threadIdx.x] = ss;
    __syncthreads();
    for (int o = 128; o > 0; o >>= 1) {
        if (threadIdx.x < o) {
            sh[threadIdx.x]  += sh[threadIdx.x + o];
            sh2[threadIdx.x] += sh2[threadIdx.x + o];
        }
        __syncthreads();
    }
    if (threadIdx.x == 0) {
        float m   = sh[0] / (float)Mrows;
        float var = sh2[0] / (float)Mrows - m * m;
        mean[col] = m;
        rstd[col] = rsqrtf(var + 1e-5f);
    }
}

__global__ __launch_bounds__(256)
void ln_relu(float* __restrict__ h1, const float* __restrict__ mean,
             const float* __restrict__ rstd, const float* __restrict__ gamma,
             const float* __restrict__ beta, int total)
{
    int i = blockIdx.x * 256 + threadIdx.x;
    if (i >= total) return;
    int col = i & 511;
    float v = (h1[i] - mean[col]) * rstd[col] * gamma[col] + beta[col];
    h1[i] = fmaxf(v, 0.f);
}

// mean-pool 100 nodes per graph -> (256, 512)
__global__ __launch_bounds__(512)
void pool100(const float* __restrict__ xm, float* __restrict__ xg)
{
    int b = blockIdx.x, d = threadIdx.x;
    float s = 0.f;
    for (int p = 0; p < 100; ++p) s += xm[((size_t)b * 100 + p) * 512 + d];
    xg[b * 512 + d] = s * 0.01f;
}

// alpha weights: alw[l,b,k] = (s@Ws.T + pre)[...] normalized by sum over k
__global__ __launch_bounds__(256)
void attn_weights(const float* __restrict__ c0, const float* __restrict__ W_alm,
                  const float* __restrict__ pre, float* __restrict__ alw)
{
    __shared__ float t1[100];
    int l = blockIdx.x;
    for (int k = threadIdx.x; k < 100; k += 256) {
        float s = 0.f;
        for (int j = 0; j < 512; ++j) s += c0[l * 512 + j] * W_alm[(size_t)k * 51712 + j];
        t1[k] = s;
    }
    __syncthreads();
    int b = threadIdx.x;           // 256 threads = 256 graphs
    float den = 0.f;
    for (int k = 0; k < 100; ++k) den += t1[k] + pre[b * 100 + k];
    float inv = 1.0f / den;
    for (int k = 0; k < 100; ++k)
        alw[(l * 256 + b) * 100 + k] = (t1[k] + pre[b * 100 + k]) * inv;
}

// new_c[l,d] = sum_i alw[l,i] * xc[i,d]   (i over 25600)
__global__ __launch_bounds__(512)
void attn_context(const float* __restrict__ alw, const float* __restrict__ xc,
                  float* __restrict__ new_c)
{
    int l = blockIdx.x, d = threadIdx.x;
    float acc = 0.f;
    for (int i = 0; i < 25600; ++i)
        acc += alw[l * 25600 + i] * xc[(size_t)i * 512 + d];
    new_c[l * 512 + d] = acc;
}

// ---------------------------------------------------------------------------
// Fused 2-layer sequential LSTM, 256 steps, batch 1, R=512, single workgroup.
// ---------------------------------------------------------------------------
__device__ __forceinline__ void lstm_gates(const float* __restrict__ Wih,
                                           const float* __restrict__ Whh,
                                           const float* __restrict__ bih,
                                           const float* __restrict__ bhh,
                                           const float* sx, const float* sh,
                                           float* sg, int t)
{
    #pragma unroll
    for (int gi = 0; gi < 2; ++gi) {
        int j = t + gi * 1024;
        const float4* wi = (const float4*)(Wih + (size_t)j * 512);
        const float4* wh = (const float4*)(Whh + (size_t)j * 512);
        float acc = bih[j] + bhh[j];
        #pragma unroll 4
        for (int k = 0; k < 128; ++k) {
            float4 a = wi[k], b = wh[k];
            float4 xv = *(const float4*)(sx + 4 * k);
            float4 hv = *(const float4*)(sh + 4 * k);
            acc += a.x * xv.x + a.y * xv.y + a.z * xv.z + a.w * xv.w
                 + b.x * hv.x + b.y * hv.y + b.z * hv.z + b.w * hv.w;
        }
        sg[j] = acc;
    }
}

__global__ __launch_bounds__(1024)
void lstm_fused(const float* __restrict__ x_emb, const float* __restrict__ h0,
                const float* __restrict__ cini,
                const float* __restrict__ Wih0, const float* __restrict__ Whh0,
                const float* __restrict__ bih0, const float* __restrict__ bhh0,
                const float* __restrict__ Wih1, const float* __restrict__ Whh1,
                const float* __restrict__ bih1, const float* __restrict__ bhh1,
                float* __restrict__ outs)
{
    __shared__ __align__(16) float sx[512], sh0[512], sc0[512], sh1[512], sc1[512];
    __shared__ __align__(16) float sg[2048];
    int t = threadIdx.x;
    if (t < 512) {
        sh0[t] = h0[t];        sc0[t] = cini[t];
        sh1[t] = h0[512 + t];  sc1[t] = cini[512 + t];
    }
    __syncthreads();
    for (int step = 0; step < 256; ++step) {
        if (t < 512) sx[t] = x_emb[step * 512 + t];
        __syncthreads();
        lstm_gates(Wih0, Whh0, bih0, bhh0, sx, sh0, sg, t);
        __syncthreads();
        if (t < 512) {
            float c = sigf(sg[512 + t]) * sc0[t] + sigf(sg[t]) * tanhf(sg[1024 + t]);
            float h = sigf(sg[1536 + t]) * tanhf(c);
            sc0[t] = c; sh0[t] = h; sx[t] = h;      // h0 out feeds layer 1
        }
        __syncthreads();
        lstm_gates(Wih1, Whh1, bih1, bhh1, sx, sh1, sg, t);
        __syncthreads();
        if (t < 512) {
            float c = sigf(sg[512 + t]) * sc1[t] + sigf(sg[t]) * tanhf(sg[1024 + t]);
            float h = sigf(sg[1536 + t]) * tanhf(c);
            sc1[t] = c; sh1[t] = h;
            outs[step * 512 + t] = h;
        }
        __syncthreads();
    }
}

// ---------------------------------------------------------------------------
extern "C" void kernel_launch(void* const* d_in, const int* in_sizes, int n_in,
                              void* d_out, int out_size, void* d_ws, size_t ws_size,
                              hipStream_t stream)
{
    const float* x      = (const float*)d_in[0];
    const int*   ei     = (const int*)  d_in[1];
    const float* h0     = (const float*)d_in[4];
    const float* c0     = (const float*)d_in[5];
    const float* W_gnn  = (const float*)d_in[6];
    const float* W_root = (const float*)d_in[7];
    const float* W_nbr  = (const float*)d_in[8];
    const float* b_conv = (const float*)d_in[9];
    const float* W1     = (const float*)d_in[10];
    const float* b1     = (const float*)d_in[11];
    const float* gamma  = (const float*)d_in[12];
    const float* beta   = (const float*)d_in[13];
    const float* W2     = (const float*)d_in[14];
    const float* b2     = (const float*)d_in[15];
    const float* W_alm  = (const float*)d_in[16];
    const float* b_alm  = (const float*)d_in[17];
    const float* Wih0   = (const float*)d_in[18];
    const float* Whh0   = (const float*)d_in[19];
    const float* bih0   = (const float*)d_in[20];
    const float* bhh0   = (const float*)d_in[21];
    const float* Wih1   = (const float*)d_in[22];
    const float* Whh1   = (const float*)d_in[23];
    const float* bih1   = (const float*)d_in[24];
    const float* bhh1   = (const float*)d_in[25];
    const float* W_fc   = (const float*)d_in[26];
    const float* b_fc   = (const float*)d_in[27];
    float* out = (float*)d_out;

    const int Nn = 25600, E = 409600;
    char* wsb = (char*)d_ws;
    size_t off = 0;
    auto take = [&](size_t bytes) -> void* {
        void* p = wsb + off;
        off += (bytes + 255) & ~(size_t)255;
        return p;
    };
    float* xg    = (float*)take((size_t)Nn * 256 * 4);
    float* aggf  = (float*)take((size_t)Nn * 256 * 4);
    float* aggb  = (float*)take((size_t)Nn * 256 * 4);
    float* xc    = (float*)take((size_t)Nn * 512 * 4);
    float* h1    = (float*)take((size_t)Nn * 512 * 4);
    float* meanb = (float*)take(512 * 4);
    float* rstdb = (float*)take(512 * 4);
    float* x_g   = (float*)take(256 * 512 * 4);
    float* pre   = (float*)take(256 * 100 * 4);
    float* alw   = (float*)take(2 * 25600 * 4);
    float* ncb   = (float*)take(2 * 512 * 4);
    float* outs1 = (float*)take(256 * 512 * 4);
    float* xmlp  = aggf;   // reuse aggf+aggb (contiguous 52.4MB) after conv

    dim3 blk(256);

    // 1) xg = x @ W_gnn             (25600 x 128 x 256)
    wmma_gemm<<<dim3(256 / NT, (Nn + MT - 1) / MT), blk, 0, stream>>>(
        x, 128, W_gnn, 256, 0, nullptr, nullptr, nullptr, xg, 256, Nn, 256, 128, 0);

    // 2) scatter-add over edges (both directions)
    hipMemsetAsync(aggf, 0, 2 * (size_t)Nn * 256 * 4, stream);
    scatter_edges<<<(E * 64) / 256, blk, 0, stream>>>(xg, ei, aggf, aggb, E);

    // 3) xc[:, :256] = relu(xg@W_root + aggf@W_nbr + b_conv); xc[:,256:] same with aggb
    dim3 gConv(256 / NT, (Nn + MT - 1) / MT);
    wmma_gemm<<<gConv, blk, 0, stream>>>(xg, 256, W_root, 256, 0, aggf, W_nbr,
                                         b_conv, xc, 512, Nn, 256, 256, 1);
    wmma_gemm<<<gConv, blk, 0, stream>>>(xg, 256, W_root, 256, 0, aggb, W_nbr,
                                         b_conv, xc + 256, 512, Nn, 256, 256, 1);

    // 4) h1 = xc @ W1 + b1          (25600 x 512 x 512)
    dim3 g512(512 / NT, (Nn + MT - 1) / MT);
    wmma_gemm<<<g512, blk, 0, stream>>>(xc, 512, W1, 512, 0, nullptr, nullptr,
                                        b1, h1, 512, Nn, 512, 512, 0);

    // 5) feature-wise norm over axis 0, then ReLU (in place)
    col_stats<<<512, blk, 0, stream>>>(h1, meanb, rstdb, Nn);
    ln_relu<<<(Nn * 512) / 256, blk, 0, stream>>>(h1, meanb, rstdb, gamma, beta, Nn * 512);

    // 6) x_mlp = relu(h1) @ W2 + b2
    wmma_gemm<<<g512, blk, 0, stream>>>(h1, 512, W2, 512, 0, nullptr, nullptr,
                                        b2, xmlp, 512, Nn, 512, 512, 0);

    // 7) per-graph mean pool -> x_g (256 x 512)
    pool100<<<256, 512, 0, stream>>>(xmlp, x_g);

    // 8) attention: pre = xc.reshape(256,51200) @ Wh.T + b_alm  (transB GEMM)
    wmma_gemm<<<dim3(1, 256 / MT), blk, 0, stream>>>(
        xc, 51200, W_alm + 512, 51712, 1, nullptr, nullptr, b_alm,
        pre, 100, 256, 100, 51200, 0);
    attn_weights<<<2, blk, 0, stream>>>(c0, W_alm, pre, alw);
    attn_context<<<2, 512, 0, stream>>>(alw, xc, ncb);

    // 9) fused 2-layer sequential LSTM (256 steps)
    lstm_fused<<<1, 1024, 0, stream>>>(x_g, h0, ncb, Wih0, Whh0, bih0, bhh0,
                                       Wih1, Whh1, bih1, bhh1, outs1);

    // 10) preds = outs1 @ W_fc + b_fc   (256 x 512 x 32000) -> d_out
    wmma_gemm<<<dim3(32000 / NT, 256 / MT), blk, 0, stream>>>(
        outs1, 512, W_fc, 32000, 0, nullptr, nullptr, b_fc,
        out, 32000, 256, 32000, 512, 0);
}
